// Part4_BahdanauAttention_Decoder_71356586655986
// MI455X (gfx1250) — matrix operand order repaired and checked by reading, not compiled, not run on hardware
//
#include <hip/hip_runtime.h>

// ---------- types ----------
typedef __attribute__((ext_vector_type(16))) __bf16 v16bf;
typedef __attribute__((ext_vector_type(8)))  float  v8f;

// fp32 -> bf16, round-to-nearest-even
static __device__ __forceinline__ __bf16 f2bf(float f) {
    union { float f; unsigned u; } v; v.f = f;
    unsigned r = v.u + 0x7FFFu + ((v.u >> 16) & 1u);
    unsigned short h = (unsigned short)(r >> 16);
    return __builtin_bit_cast(__bf16, h);
}

static __device__ __forceinline__ float sigmoidf_(float x) {
    return 1.0f / (1.0f + __expf(-x));
}

// ======================================================================
// Weight repack: fp32 [K,N] row-major -> bf16 WMMA B-fragment order.
// Tile = (nt, kt) of 32x16 (KxN). tileIdx = nt*KT + kt (n-major).
// Within a tile: lane L holds column n = L&15, K = (j>=8?16:0)+(L>=16?8:0)+(j&7), j=0..15,
// stored as 16 contiguous halves -> one 32B load per lane in the GEMM.
// ======================================================================
__global__ void prep_bfrag(const float* __restrict__ W, __bf16* __restrict__ dst,
                           int K, int N) {
    size_t idx = (size_t)blockIdx.x * 256 + threadIdx.x;
    size_t total = (size_t)K * N;
    if (idx >= total) return;
    int tile   = (int)(idx >> 9);
    int within = (int)(idx & 511);
    int ln = within >> 4;
    int jj = within & 15;
    int KT = K >> 5;
    int nt = tile / KT;
    int kt = tile - nt * KT;
    int k = kt * 32 + ((jj >> 3) << 4) + ((ln >> 4) << 3) + (jj & 7);
    int n = nt * 16 + (ln & 15);
    dst[idx] = f2bf(W[(size_t)k * N + n]);
}

// ======================================================================
// A-fragment staging mapping (16x32 bf16 A tile, ISA 7.12.2):
// element (r, k): khi2 = k>>3; lane = r + ((khi2&1)<<4); j = (k&7) + ((khi2>>1)<<3)
// ======================================================================

// Generic GEMM: C[M,N] = (beta?C:0) + A[M,K]@Bfrag + bias0 + bias1
// Workgroup = 8 waves x 32 lanes; one 16-row M tile per workgroup (blockIdx.y);
// each wave computes NT consecutive 16-col N tiles. K chunked through LDS.
#define GWAVES 8
template<int NT>
__global__ void gemm_bias(const float* __restrict__ A, int lda,
                          const __bf16* __restrict__ Bf,
                          float* __restrict__ C, int ldc,
                          int N, int K,
                          const float* __restrict__ bias0,
                          const float* __restrict__ bias1,
                          int beta) {
    const int KC = 512;
    __shared__ __align__(32) __bf16 ldsA[16 * KC];   // 16 KB, frag-ordered
    const int tid  = threadIdx.x;
    const int lane = tid & 31;
    const int wave = tid >> 5;
    const int m0 = blockIdx.y * 16;
    const int NTILES = N >> 4;
    const int KT_TOT = K >> 5;
    const int ntBase = (blockIdx.x * GWAVES + wave) * NT;

    v8f acc[NT] = {};

    for (int k0 = 0; k0 < K; k0 += KC) {
        int kc = (K - k0 < KC) ? (K - k0) : KC;
        __syncthreads();
        // cooperative fp32->bf16 A staging in fragment order
        #pragma unroll 4
        for (int e = tid; e < 16 * kc; e += 256) {
            int r  = e & 15;
            int kg = e >> 4;              // K index within chunk
            int kt = kg >> 5;
            int kk = kg & 31;
            int khi2 = kk >> 3;
            int ln = r + ((khi2 & 1) << 4);
            int jj = (kk & 7) + ((khi2 >> 1) << 3);
            ldsA[kt * 512 + ln * 16 + jj] = f2bf(A[(size_t)(m0 + r) * lda + k0 + kg]);
        }
        __syncthreads();
        int ktn = kc >> 5;
        int ktG = k0 >> 5;
        #pragma unroll 1
        for (int kt = 0; kt < ktn; ++kt) {
            v16bf a = *(const v16bf*)(&ldsA[kt * 512 + lane * 16]);
            #pragma unroll
            for (int j = 0; j < NT; ++j) {
                int nt = ntBase + j;
                if (nt < NTILES) {                      // uniform branch (EXEC intact)
                    const __bf16* bp = Bf + ((size_t)nt * KT_TOT + ktG + kt) * 512 + lane * 16;
                    v16bf b = *(const v16bf*)bp;
                    acc[j] = __builtin_amdgcn_wmma_f32_16x16x32_bf16(
                        false, a, false, b, (short)0, acc[j], false, false);
                }
            }
        }
    }
    // store: C/D layout - VGPR r: lanes 0-15 row r, lanes 16-31 row r+8
    const int rowHalf = (lane >> 4) << 3;
    const int col = lane & 15;
    #pragma unroll
    for (int j = 0; j < NT; ++j) {
        int nt = ntBase + j;
        if (nt >= NTILES) continue;
        int n = nt * 16 + col;
        float bsum = (bias0 ? bias0[n] : 0.0f) + (bias1 ? bias1[n] : 0.0f);
        #pragma unroll
        for (int r = 0; r < 8; ++r) {
            size_t off = (size_t)(m0 + r + rowHalf) * ldc + n;
            float v = acc[j][r] + bsum;
            if (beta) v += C[off];
            C[off] = v;
        }
    }
}

// ======================================================================
// Fused attention score:
//   score[s,b] = sum_n tanh( (enc@W_e)[s,b,n] + hp[b,n] ) * w_a[n] + b_a
// One workgroup per 16-row (s,b) tile; enc rows staged once to 32KB LDS in
// A-frag order; 8 waves x 8 N-tiles each. kt is OUTER so one A-fragment
// feeds 8 WMMAs; 8 accumulators (64 VGPRs) stay resident. 2048 WMMA / WG.
// ======================================================================
__global__ void attn_score(const float* __restrict__ enc,   // [S*B, 1024]
                           const float* __restrict__ hp,    // [128, 1024]
                           const __bf16* __restrict__ Bf,   // frag W_e, KT=32, NT=64
                           const float* __restrict__ wa,    // [1024]
                           const float* __restrict__ ba,    // scalar
                           float* __restrict__ score) {     // [S*B]
    __shared__ __align__(32) __bf16 ldsA[16 * 1024];        // 32 KB
    __shared__ float ssum[16];
    const int tid  = threadIdx.x;
    const int lane = tid & 31;
    const int wave = tid >> 5;
    const int m0 = blockIdx.x * 16;

    #pragma unroll 4
    for (int e = tid; e < 16 * 1024; e += 256) {
        int r  = e & 15;
        int kg = e >> 4;
        int kt = kg >> 5;
        int kk = kg & 31;
        int khi2 = kk >> 3;
        int ln = r + ((khi2 & 1) << 4);
        int jj = (kk & 7) + ((khi2 >> 1) << 3);
        ldsA[kt * 512 + ln * 16 + jj] = f2bf(enc[(size_t)(m0 + r) * 1024 + kg]);
    }
    if (tid < 16) ssum[tid] = 0.0f;
    __syncthreads();

    const int rowHalf = (lane >> 4) << 3;
    const int col = lane & 15;
    const int bRow0 = m0 & 127;      // b of row 0 (tiles never straddle s: 128%16==0)

    // 8 N-tile accumulators stay live across the K loop (64 VGPRs)
    v8f acc[8] = {};
    const __bf16* bbase = Bf + ((size_t)(wave * 8) * 32) * 512 + lane * 16;

    #pragma unroll 1
    for (int kt = 0; kt < 32; ++kt) {
        v16bf a = *(const v16bf*)(&ldsA[kt * 512 + lane * 16]);
        const __bf16* bp = bbase + (size_t)kt * 512;
        #pragma unroll
        for (int j = 0; j < 8; ++j) {
            v16bf b = *(const v16bf*)(bp + (size_t)j * (32 * 512));
            acc[j] = __builtin_amdgcn_wmma_f32_16x16x32_bf16(
                false, a, false, b, (short)0, acc[j], false, false);
        }
    }

    // tanh + weighted column sum, accumulated per (row, lane) partial
    float rowAcc[8] = {0,0,0,0,0,0,0,0};
    #pragma unroll
    for (int j = 0; j < 8; ++j) {
        int n = (wave * 8 + j) * 16 + col;
        float wv = wa[n];
        #pragma unroll
        for (int r = 0; r < 8; ++r) {
            int bidx = bRow0 + r + rowHalf;
            rowAcc[r] += tanhf(acc[j][r] + hp[(size_t)bidx * 1024 + n]) * wv;
        }
    }
    // reduce over the 16 columns (lanes) inside each half-wave
    #pragma unroll
    for (int r = 0; r < 8; ++r) {
        float v = rowAcc[r];
        v += __shfl_xor(v, 1, 32);
        v += __shfl_xor(v, 2, 32);
        v += __shfl_xor(v, 4, 32);
        v += __shfl_xor(v, 8, 32);
        rowAcc[r] = v;
    }
    if (col == 0) {
        #pragma unroll
        for (int r = 0; r < 8; ++r) atomicAdd(&ssum[r + rowHalf], rowAcc[r]);
    }
    __syncthreads();
    if (tid < 16) score[m0 + tid] = ssum[tid] + ba[0];
}

// softmax over S=256 for each column b of score[S,B]
__global__ void softmax_col(float* __restrict__ score) {
    __shared__ float red[256];
    int b = blockIdx.x;
    int s = threadIdx.x;
    float v = score[(size_t)s * 128 + b];
    red[s] = v; __syncthreads();
    for (int off = 128; off > 0; off >>= 1) {
        if (s < off) red[s] = fmaxf(red[s], red[s + off]);
        __syncthreads();
    }
    float mx = red[0]; __syncthreads();
    float e = __expf(v - mx);
    red[s] = e; __syncthreads();
    for (int off = 128; off > 0; off >>= 1) {
        if (s < off) red[s] += red[s + off];
        __syncthreads();
    }
    score[(size_t)s * 128 + b] = e / red[0];
}

// ctx[b,h] = sum_s w[s,b] * enc[s,b,h], written into x at a column offset
__global__ void context_k(const float* __restrict__ w,
                          const float* __restrict__ enc,
                          float* __restrict__ xout, int ldx) {
    int b = blockIdx.x;
    int h = blockIdx.y * 256 + threadIdx.x;
    float acc = 0.0f;
    for (int s = 0; s < 256; ++s)
        acc += w[(size_t)s * 128 + b] * enc[((size_t)s * 128 + b) * 1024 + h];
    xout[(size_t)b * ldx + h] = acc;
}

// x[:,0:512) = emb[ids], x[:,2560:3456) = env   (ctx fills 512..2560)
__global__ void build_x(const int* __restrict__ ids, const float* __restrict__ emb,
                        const float* __restrict__ env, float* __restrict__ x) {
    const int E = 512, ENV = 896, IN = 3456;
    int idx = blockIdx.x * 256 + threadIdx.x;
    int totalE = 128 * E;
    if (idx < totalE) {
        int b = idx / E, e = idx - b * E;
        x[(size_t)b * IN + e] = emb[(size_t)ids[b] * E + e];
    } else {
        int k = idx - totalE;
        if (k < 128 * ENV) {
            int b = k / ENV, j = k - b * ENV;
            x[(size_t)b * IN + 2560 + j] = env[(size_t)b * ENV + j];
        }
    }
}

// LSTM cell elementwise (torch gate order i,f,g,o)
__global__ void lstm_cell(const float* __restrict__ gates, const float* __restrict__ c0,
                          float* __restrict__ h1, float* __restrict__ c1) {
    int idx = blockIdx.x * 256 + threadIdx.x;   // B*H
    int b = idx >> 10, n = idx & 1023;
    const float* g = gates + (size_t)b * 4096;
    float i  = sigmoidf_(g[n]);
    float f  = sigmoidf_(g[1024 + n]);
    float gg = tanhf(g[2048 + n]);
    float o  = sigmoidf_(g[3072 + n]);
    float c  = f * c0[idx] + i * gg;
    c1[idx] = c;
    h1[idx] = o * tanhf(c);
}

// ======================================================================
extern "C" void kernel_launch(void* const* d_in, const int* in_sizes, int n_in,
                              void* d_out, int out_size, void* d_ws, size_t ws_size,
                              hipStream_t stream) {
    const int B = 128, S = 256, H = 1024, E = 512, ENV = 896, V = 10000;
    const int IN = E + H + H + ENV;  // 3456
    const int G4 = 4 * H;            // 4096

    const int*   ids   = (const int*)  d_in[0];
    const float* prevO = (const float*)d_in[1];
    const float* encO  = (const float*)d_in[2];
    const float* env   = (const float*)d_in[3];
    const float* h0    = (const float*)d_in[4];
    const float* c0    = (const float*)d_in[5];
    const float* emb   = (const float*)d_in[6];
    const float* W_hp  = (const float*)d_in[7];  const float* b_hp = (const float*)d_in[8];
    const float* W_ep  = (const float*)d_in[9];  const float* b_ep = (const float*)d_in[10];
    const float* w_ap  = (const float*)d_in[11]; const float* b_ap = (const float*)d_in[12];
    const float* W_hc  = (const float*)d_in[13]; const float* b_hc = (const float*)d_in[14];
    const float* W_ec  = (const float*)d_in[15]; const float* b_ec = (const float*)d_in[16];
    const float* w_ac  = (const float*)d_in[17]; const float* b_ac = (const float*)d_in[18];
    const float* W_ih  = (const float*)d_in[19]; const float* b_ih = (const float*)d_in[20];
    const float* W_hh  = (const float*)d_in[21]; const float* b_hh = (const float*)d_in[22];
    const float* W_out = (const float*)d_in[23]; const float* b_out= (const float*)d_in[24];

    // ---- workspace carve-out (≈71 MB) ----
    char* wsp = (char*)d_ws;
    auto carve = [&](size_t bytes) -> char* {
        char* p = wsp; wsp += (bytes + 255) & ~(size_t)255; return p;
    };
    __bf16* wf_ep  = (__bf16*)carve((size_t)H * H * 2);
    __bf16* wf_ec  = (__bf16*)carve((size_t)H * H * 2);
    __bf16* wf_hp  = (__bf16*)carve((size_t)H * H * 2);
    __bf16* wf_hc  = (__bf16*)carve((size_t)H * H * 2);
    __bf16* wf_ih  = (__bf16*)carve((size_t)IN * G4 * 2);
    __bf16* wf_hh  = (__bf16*)carve((size_t)H * G4 * 2);
    __bf16* wf_out = (__bf16*)carve((size_t)H * V * 2);
    float* hp      = (float*)carve((size_t)B * H * 4);
    float* hc      = (float*)carve((size_t)B * H * 4);
    float* score_p = (float*)carve((size_t)S * B * 4);
    float* score_c = (float*)carve((size_t)S * B * 4);
    float* x       = (float*)carve((size_t)B * IN * 4);
    float* gates   = (float*)carve((size_t)B * G4 * 4);

    float* outp = (float*)d_out;                       // [B,V]
    float* h1   = outp + (size_t)B * V;                // [B,H]
    float* c1   = h1 + (size_t)B * H;                  // [B,H]

    auto prep = [&](const float* W, __bf16* dst, int K, int N) {
        size_t total = (size_t)K * N;
        prep_bfrag<<<dim3((unsigned)((total + 255) / 256)), 256, 0, stream>>>(W, dst, K, N);
    };
    // one-time weight repacks
    prep(W_ep, wf_ep, H, H);
    prep(W_ec, wf_ec, H, H);
    prep(W_hp, wf_hp, H, H);
    prep(W_hc, wf_hc, H, H);
    prep(W_ih, wf_ih, IN, G4);
    prep(W_hh, wf_hh, H, G4);
    prep(W_out, wf_out, H, V);

    auto gemm = [&](const float* A, int lda, const __bf16* Bf, float* C, int ldc,
                    int M, int N, int K, const float* bs0, const float* bs1, int beta) {
        int ntiles = N >> 4;
        int gx = (ntiles + GWAVES * 4 - 1) / (GWAVES * 4);
        gemm_bias<4><<<dim3(gx, M >> 4), 256, 0, stream>>>(A, lda, Bf, C, ldc, N, K, bs0, bs1, beta);
    };

    // hp = h0@W_hp + b_hp + b_ep ; hc = h0@W_hc + b_hc + b_ec
    gemm(h0, H, wf_hp, hp, H, B, H, H, b_hp, b_ep, 0);
    gemm(h0, H, wf_hc, hc, H, B, H, H, b_hc, b_ec, 0);

    // fused attention scores (the 137-GFLOP hot path)
    attn_score<<<(S * B) / 16, 256, 0, stream>>>(prevO, hp, wf_ep, w_ap, b_ap, score_p);
    attn_score<<<(S * B) / 16, 256, 0, stream>>>(encO,  hc, wf_ec, w_ac, b_ac, score_c);

    softmax_col<<<B, S, 0, stream>>>(score_p);
    softmax_col<<<B, S, 0, stream>>>(score_c);

    // contexts written directly into the concat buffer x
    context_k<<<dim3(B, H / 256), 256, 0, stream>>>(score_p, prevO, x + E,     IN);
    context_k<<<dim3(B, H / 256), 256, 0, stream>>>(score_c, encO,  x + E + H, IN);
    build_x<<<(B * (E + ENV) + 255) / 256, 256, 0, stream>>>(ids, emb, env, x);

    // gates = x@W_ih + b_ih + b_hh ; gates += h0@W_hh
    gemm(x,  IN, wf_ih, gates, G4, B, G4, IN, b_ih, b_hh, 0);
    gemm(h0, H,  wf_hh, gates, G4, B, G4, H,  nullptr, nullptr, 1);

    lstm_cell<<<(B * H) / 256, 256, 0, stream>>>(gates, c0, h1, c1);

    // out = h1 @ W_out + b_out  (N = 10000 = 625 full 16-col tiles)
    gemm(h1, H, wf_out, outp, V, B, V, H, b_out, nullptr, 0);
}
